// TopDownCoreHC_24232205484543
// MI455X (gfx1250) — matrix operand order repaired and checked by reading, not compiled, not run on hardware
//
#include <hip/hip_runtime.h>

#define B_  512
#define S_  36
#define H_  1024
#define AH_ 512
#define P_  6
#define E_  1024

typedef __attribute__((ext_vector_type(16))) __bf16 v16bf;
typedef __attribute__((ext_vector_type(8)))  float  v8f;

__device__ __forceinline__ float sigmoidf_(float x) { return 1.0f / (1.0f + __expf(-x)); }

__device__ __forceinline__ void pack8(v16bf& d, int base, float4 a, float4 b) {
    d[base + 0] = (__bf16)a.x; d[base + 1] = (__bf16)a.y;
    d[base + 2] = (__bf16)a.z; d[base + 3] = (__bf16)a.w;
    d[base + 4] = (__bf16)b.x; d[base + 5] = (__bf16)b.y;
    d[base + 6] = (__bf16)b.z; d[base + 7] = (__bf16)b.w;
}

// C[M,N](ldc) = epi( A[M,K](lda) @ W[N,K](ldw)^T + bias )
// EPI: 0 store, 1 accumulate(+bias), 2 sigmoid(relu(x)), 3 relu(x), 4 x + res
// Block: 256 threads = 8 waves in 2(M) x 4(N); wave tile 32x64 (2 A-frags share
// each converted B-frag -> 6 cvt_pk and 48 B of loads per WMMA); K step 32.
template <int EPI>
__global__ __launch_bounds__(256) void wmma_gemm(
    const float* __restrict__ A, int lda,
    const float* __restrict__ W, int ldw,
    const float* __restrict__ bias,
    const float* __restrict__ res,
    float* __restrict__ C, int ldc,
    int M, int N, int K)
{
    const int tid  = threadIdx.x;
    const int wave = tid >> 5;
    const int lane = tid & 31;
    const int wm   = wave >> 2;                 // 0..1
    const int wn   = wave & 3;                  // 0..3
    const int m0   = blockIdx.y * 64 + wm * 32;
    const int n0   = blockIdx.x * 256 + wn * 64;
    if (m0 >= M || n0 >= N) return;

    const int hsel = lane >> 4;                 // 0: lanes 0-15, 1: lanes 16-31
    const int l15  = lane & 15;

    v8f acc[2][4];
#pragma unroll
    for (int i = 0; i < 2; ++i)
#pragma unroll
        for (int j = 0; j < 4; ++j) acc[i][j] = (v8f)0.0f;

    const float* Arow0 = A + (size_t)(m0 + l15) * lda;
    const float* Arow1 = A + (size_t)(m0 + 16 + l15) * lda;

    for (int kk = 0; kk < K; kk += 32) {
        // A fragments (16x32 bf16): lanes<16 hold K kk+0..7 / kk+16..23, lanes>=16 +8
        const int ka = kk + hsel * 8;
        v16bf af[2];
        {
            float4 a0 = *(const float4*)(Arow0 + ka);
            float4 a1 = *(const float4*)(Arow0 + ka + 4);
            float4 a2 = *(const float4*)(Arow0 + ka + 16);
            float4 a3 = *(const float4*)(Arow0 + ka + 20);
            pack8(af[0], 0, a0, a1);
            pack8(af[0], 8, a2, a3);
            float4 b0 = *(const float4*)(Arow1 + ka);
            float4 b1 = *(const float4*)(Arow1 + ka + 4);
            float4 b2 = *(const float4*)(Arow1 + ka + 16);
            float4 b3 = *(const float4*)(Arow1 + ka + 20);
            pack8(af[1], 0, b0, b1);
            pack8(af[1], 8, b2, b3);
        }
#pragma unroll
        for (int j = 0; j < 4; ++j) {
            // B fragment (32x16): lane column = l15, K kk+0..7 / kk+16..23, lanes>=16 +8
            const float* Wr = W + (size_t)(n0 + j * 16 + l15) * ldw + kk + hsel * 8;
            float4 b0 = *(const float4*)(Wr);
            float4 b1 = *(const float4*)(Wr + 4);
            float4 b2 = *(const float4*)(Wr + 16);
            float4 b3 = *(const float4*)(Wr + 20);
            v16bf wf;
            pack8(wf, 0, b0, b1);
            pack8(wf, 8, b2, b3);
            acc[0][j] = __builtin_amdgcn_wmma_f32_16x16x32_bf16(
                false, af[0], false, wf, (short)0, acc[0][j], false, false);
            acc[1][j] = __builtin_amdgcn_wmma_f32_16x16x32_bf16(
                false, af[1], false, wf, (short)0, acc[1][j], false, false);
        }
    }

#pragma unroll
    for (int i = 0; i < 2; ++i) {
#pragma unroll
        for (int j = 0; j < 4; ++j) {
            const int col = n0 + j * 16 + l15;
            const float bv = bias ? bias[col] : 0.0f;
#pragma unroll
            for (int v = 0; v < 8; ++v) {
                const int row = m0 + i * 16 + hsel * 8 + v;  // C/D: VGPR v -> M=v(+8)
                const size_t idx = (size_t)row * ldc + col;
                float x = acc[i][j][v] + bv;
                if      (EPI == 0) C[idx] = x;
                else if (EPI == 1) C[idx] += x;
                else if (EPI == 2) C[idx] = sigmoidf_(fmaxf(x, 0.0f));
                else if (EPI == 3) C[idx] = fmaxf(x, 0.0f);
                else               C[idx] = x + res[idx];
            }
        }
    }
}

// One block per batch row: scores -> softmax -> mask renorm -> weighted feat sum.
__global__ __launch_bounds__(256) void attention_kernel(
    const float* __restrict__ atth,    // [B, AH]
    const float* __restrict__ alphaW,  // [AH]
    const float* __restrict__ alphab,  // [1]
    const float* __restrict__ pfeats,  // [B, S, AH]
    const float* __restrict__ feats,   // [B, S, H]
    const float* __restrict__ mask,    // [B, S] or null
    float* __restrict__ out)           // [B, H]
{
    __shared__ float s_h[AH_];
    __shared__ float s_sc[S_];
    __shared__ float s_w[S_];
    const int b    = blockIdx.x;
    const int tid  = threadIdx.x;
    const int wave = tid >> 5;
    const int lane = tid & 31;

    for (int a = tid; a < AH_; a += 256) s_h[a] = atth[(size_t)b * AH_ + a];
    __syncthreads();

    for (int s = wave; s < S_; s += 8) {
        const float* p = pfeats + ((size_t)b * S_ + s) * AH_;
        float part = 0.0f;
        for (int a = lane; a < AH_; a += 32)
            part += tanhf(p[a] + s_h[a]) * alphaW[a];
        for (int off = 16; off > 0; off >>= 1)
            part += __shfl_xor(part, off, 32);
        if (lane == 0) s_sc[s] = part + alphab[0];
    }
    __syncthreads();

    if (tid == 0) {
        float mx = s_sc[0];
        for (int s = 1; s < S_; ++s) mx = fmaxf(mx, s_sc[s]);
        float sum = 0.0f;
        for (int s = 0; s < S_; ++s) { float e = __expf(s_sc[s] - mx); s_w[s] = e; sum += e; }
        float inv = 1.0f / sum;
        if (mask) {
            float msum = 0.0f;
            for (int s = 0; s < S_; ++s) { s_w[s] = s_w[s] * inv * mask[(size_t)b * S_ + s]; msum += s_w[s]; }
            float inv2 = 1.0f / (msum + 1e-10f);
            for (int s = 0; s < S_; ++s) s_w[s] *= inv2;
        } else {
            for (int s = 0; s < S_; ++s) s_w[s] *= inv;
        }
    }
    __syncthreads();

    for (int d = tid; d < H_; d += 256) {
        const float* f = feats + (size_t)b * S_ * H_ + d;
        float acc = 0.0f;
#pragma unroll 4
        for (int s = 0; s < S_; ++s) acc += s_w[s] * f[(size_t)s * H_];
        out[(size_t)b * H_ + d] = acc;
    }
}

__global__ void lstm_pointwise(const float* __restrict__ gates, const float* __restrict__ cprev,
                               float* __restrict__ hout, float* __restrict__ hout2,
                               float* __restrict__ cout)
{
    int i = blockIdx.x * blockDim.x + threadIdx.x;
    if (i >= B_ * H_) return;
    int b = i / H_, d = i - b * H_;
    const float* g = gates + (size_t)b * 4 * H_;
    float ig = sigmoidf_(g[d]);
    float fg = sigmoidf_(g[H_ + d]);
    float gg = tanhf(g[2 * H_ + d]);
    float og = sigmoidf_(g[3 * H_ + d]);
    float c2 = fg * cprev[i] + ig * gg;
    float h2 = og * tanhf(c2);
    cout[i] = c2;
    hout[i] = h2;
    if (hout2) hout2[i] = h2;
}

__global__ void human_mean_kernel(const float* __restrict__ hf, const float* __restrict__ bm,
                                  float* __restrict__ out)
{
    int i = blockIdx.x * blockDim.x + threadIdx.x;
    if (i >= B_ * H_) return;
    int b = i / H_, d = i - b * H_;
    float msum = 0.0f;
    for (int p = 0; p < P_; ++p) msum += bm[b * P_ + p];
    float s = 0.0f;
    for (int p = 0; p < P_; ++p) s += hf[((size_t)b * P_ + p) * H_ + d];
    out[i] = s / (msum + 1e-10f);
}

__global__ void fuse_ori_kernel(const float* __restrict__ ah, const float* __restrict__ gh,
                                const float* __restrict__ ao, const float* __restrict__ go,
                                const float* __restrict__ aori, float* __restrict__ out)
{
    int i = blockIdx.x * blockDim.x + threadIdx.x;
    if (i >= B_ * H_) return;
    out[i] = ah[i] * gh[i] + ao[i] * go[i] + aori[i];
}

__global__ void copy2d_kernel(float* __restrict__ dst, int dld, int doff,
                              const float* __restrict__ src, int sld, int soff,
                              int rows, int cols)
{
    int i = blockIdx.x * blockDim.x + threadIdx.x;
    if (i >= rows * cols) return;
    int r = i / cols, c = i - r * cols;
    dst[(size_t)r * dld + doff + c] = src[(size_t)r * sld + soff + c];
}

extern "C" void kernel_launch(void* const* d_in, const int* in_sizes, int n_in,
                              void* d_out, int out_size, void* d_ws, size_t ws_size,
                              hipStream_t stream)
{
    (void)in_sizes; (void)n_in; (void)out_size; (void)ws_size;
    auto in = [&](int i) { return (const float*)d_in[i]; };

    const float* xt          = in(0);
    const float* fc          = in(1);
    const float* attf[4]     = { in(2), in(4), in(6), in(8) };
    const float* pattf[4]    = { in(3), in(5), in(7), in(9) };
    const float* masks[3]    = { in(10), in(11), in(12) };
    const float* human_feats = in(13);
    const float* body_masks  = in(14);
    const float* state_h     = in(15);
    const float* state_c     = in(16);
    const float* W_ih_att    = in(17);
    const float* W_hh_att    = in(18);
    const float* b_ih_att    = in(19);
    const float* b_hh_att    = in(20);
    const float* W_ih_lang   = in(21);
    const float* W_hh_lang   = in(22);
    const float* b_ih_lang   = in(23);
    const float* b_hh_lang   = in(24);
    const float* h2W[4]      = { in(25), in(29), in(33), in(37) };
    const float* h2b[4]      = { in(26), in(30), in(34), in(38) };
    const float* aW[4]       = { in(27), in(31), in(35), in(39) };
    const float* ab[4]       = { in(28), in(32), in(36), in(40) };
    const float* hgW = in(41); const float* hgb = in(42);
    const float* ngW = in(43); const float* ngb = in(44);
    const float* c1W = in(45); const float* c1b = in(46);
    const float* c2W = in(47); const float* c2b = in(48);

    float* ws = (float*)d_ws;
    size_t off = 0;
    auto alloc = [&](size_t n) { float* p = ws + off; off += n; return p; };
    float* att_in   = alloc((size_t)B_ * 3 * H_);
    float* gates    = alloc((size_t)B_ * 4 * H_);
    float* atth     = alloc((size_t)B_ * AH_);
    float* att_r[4];
    for (int i = 0; i < 4; ++i) att_r[i] = alloc((size_t)B_ * H_);
    float* cat2     = alloc((size_t)B_ * 2 * H_);
    float* gate_h   = alloc((size_t)B_ * H_);
    float* gate_o   = alloc((size_t)B_ * H_);
    float* att_part = alloc((size_t)B_ * H_);
    float* cali_mid = alloc((size_t)B_ * H_);
    float* att_hum  = alloc((size_t)B_ * H_);
    float* att_ores = alloc((size_t)B_ * H_);
    float* lang_in  = alloc((size_t)B_ * 3 * H_);

    const int BH = B_ * H_;
    float* out    = (float*)d_out;
    float* h_att  = out + BH;       // stack_h[0]
    float* c_att  = out + 3 * BH;   // stack_c[0]
    float* h_lang = out;            // output (also copied to stack_h[1])
    float* c_lang = out + 4 * BH;   // stack_c[1]

    const int TPB = 256;
    auto cgrid = [&](int n) { return dim3((unsigned)((n + TPB - 1) / TPB)); };

    auto gemm = [&](int epi, const float* A, int lda, const float* W, int ldw,
                    const float* bias, const float* res, float* C, int ldc,
                    int M, int N, int K) {
        dim3 g((unsigned)(N / 256), (unsigned)(M / 64)), blk(256);
        switch (epi) {
        case 0: wmma_gemm<0><<<g, blk, 0, stream>>>(A, lda, W, ldw, bias, res, C, ldc, M, N, K); break;
        case 1: wmma_gemm<1><<<g, blk, 0, stream>>>(A, lda, W, ldw, bias, res, C, ldc, M, N, K); break;
        case 2: wmma_gemm<2><<<g, blk, 0, stream>>>(A, lda, W, ldw, bias, res, C, ldc, M, N, K); break;
        case 3: wmma_gemm<3><<<g, blk, 0, stream>>>(A, lda, W, ldw, bias, res, C, ldc, M, N, K); break;
        default: wmma_gemm<4><<<g, blk, 0, stream>>>(A, lda, W, ldw, bias, res, C, ldc, M, N, K); break;
        }
    };
    auto copy2 = [&](float* dst, int dld, int doff, const float* src, int sld, int soff,
                     int rows, int cols) {
        copy2d_kernel<<<cgrid(rows * cols), TPB, 0, stream>>>(dst, dld, doff, src, sld, soff, rows, cols);
    };

    // ---- attention LSTM: input = [state_h[1] | fc_ori | xt] ----
    copy2(att_in, 3 * H_, 0,      state_h + BH, H_,    0,      B_, H_);
    copy2(att_in, 3 * H_, H_,     fc,           3 * H_, 0,     B_, H_);
    copy2(att_in, 3 * H_, 2 * H_, xt,           E_,    0,      B_, E_);
    gemm(0, att_in, 3 * H_, W_ih_att, 3 * H_, b_ih_att, nullptr, gates, 4 * H_, B_, 4 * H_, 3 * H_);
    gemm(1, state_h, H_,    W_hh_att, H_,     b_hh_att, nullptr, gates, 4 * H_, B_, 4 * H_, H_);
    lstm_pointwise<<<cgrid(BH), TPB, 0, stream>>>(gates, state_c, h_att, nullptr, c_att);

    // ---- 4 attentions (mpm reuses the 'ori' mask) ----
    const float* amask[4] = { masks[0], masks[1], masks[2], masks[2] };
    for (int i = 0; i < 4; ++i) {
        gemm(0, h_att, H_, h2W[i], H_, h2b[i], nullptr, atth, AH_, B_, AH_, H_);
        attention_kernel<<<dim3(B_), TPB, 0, stream>>>(atth, aW[i], ab[i], pattf[i], attf[i],
                                                       amask[i], att_r[i]);
    }

    // ---- human calibration: att_human = att_activity + cali2(relu(cali1([mean|att_act]))) ----
    human_mean_kernel<<<cgrid(BH), TPB, 0, stream>>>(human_feats, body_masks, att_part);
    copy2(cat2, 2 * H_, 0,  att_part, H_, 0, B_, H_);
    copy2(cat2, 2 * H_, H_, att_r[0], H_, 0, B_, H_);
    gemm(3, cat2, 2 * H_, c1W, 2 * H_, c1b, nullptr,  cali_mid, H_, B_, H_, 2 * H_);
    gemm(4, cali_mid, H_, c2W, H_,     c2b, att_r[0], att_hum,  H_, B_, H_, H_);

    // ---- gates: sigmoid(relu([fc_x | h_att] @ W.T + b)) ----
    copy2(cat2, 2 * H_, 0,  fc,    3 * H_, H_,     B_, H_);   // fc_act
    copy2(cat2, 2 * H_, H_, h_att, H_,     0,      B_, H_);
    gemm(2, cat2, 2 * H_, hgW, 2 * H_, hgb, nullptr, gate_h, H_, B_, H_, 2 * H_);
    copy2(cat2, 2 * H_, 0,  fc,    3 * H_, 2 * H_, B_, H_);   // fc_back
    copy2(cat2, 2 * H_, H_, h_att, H_,     0,      B_, H_);
    gemm(2, cat2, 2 * H_, ngW, 2 * H_, ngb, nullptr, gate_o, H_, B_, H_, 2 * H_);

    fuse_ori_kernel<<<cgrid(BH), TPB, 0, stream>>>(att_hum, gate_h, att_r[1], gate_o,
                                                   att_r[2], att_ores);

    // ---- language LSTM: input = [att_mpm | att_ori_res | h_att] ----
    copy2(lang_in, 3 * H_, 0,      att_r[3], H_, 0, B_, H_);
    copy2(lang_in, 3 * H_, H_,     att_ores, H_, 0, B_, H_);
    copy2(lang_in, 3 * H_, 2 * H_, h_att,    H_, 0, B_, H_);
    gemm(0, lang_in, 3 * H_,  W_ih_lang, 3 * H_, b_ih_lang, nullptr, gates, 4 * H_, B_, 4 * H_, 3 * H_);
    gemm(1, state_h + BH, H_, W_hh_lang, H_,     b_hh_lang, nullptr, gates, 4 * H_, B_, 4 * H_, H_);
    lstm_pointwise<<<cgrid(BH), TPB, 0, stream>>>(gates, state_c + BH, h_lang, out + 2 * BH, c_lang);
}
